// SelfAttention_652835029618
// MI455X (gfx1250) — matrix operand order repaired and checked by reading, not compiled, not run on hardware
//
#include <hip/hip_runtime.h>

#define Nn 8192
#define Ee 512

typedef __attribute__((ext_vector_type(16))) __bf16 v16bf;
typedef __attribute__((ext_vector_type(8)))  float  v8f;

union FragBf { v16bf v; uint4 q[2]; };

__device__ __forceinline__ unsigned short f32_to_bf16_rne(float f) {
  unsigned int u = __float_as_uint(f);
  u += 0x7FFFu + ((u >> 16) & 1u);      // round-to-nearest-even
  return (unsigned short)(u >> 16);
}

// ---------------------------------------------------------------------------
// fp32 -> bf16 elementwise convert (float4 in, 4xbf16 packed u64 out)
// ---------------------------------------------------------------------------
__global__ void __launch_bounds__(256) cvt_f32_to_bf16(const float* __restrict__ in,
                                                       unsigned short* __restrict__ out,
                                                       int n4) {
  int i = blockIdx.x * 256 + threadIdx.x;
  if (i >= n4) return;
  float4 f = ((const float4*)in)[i];
  unsigned long long p =
      (unsigned long long)f32_to_bf16_rne(f.x) |
      ((unsigned long long)f32_to_bf16_rne(f.y) << 16) |
      ((unsigned long long)f32_to_bf16_rne(f.z) << 32) |
      ((unsigned long long)f32_to_bf16_rne(f.w) << 48);
  ((unsigned long long*)out)[i] = p;
}

// ---------------------------------------------------------------------------
// C = A (bf16, MxK row-major) * B (bf16, NxK row-major = math-B transposed)
//     [+ bias over N], fp32 WMMA accumulation.
// OUT_MODE: 0 = bf16 C[m*ldc+n]; 1 = bf16 transposed C[n*ldc+m]; 2 = f32 C[m*ldc+n]
// Block: 256 threads (8 wave32), tile 128x128, K-step 32, double-buffered LDS.
// Each wave: 32x64 = 2x4 v_wmma_f32_16x16x32_bf16 tiles (8 WMMA / K-step).
// ---------------------------------------------------------------------------
template <bool ADD_BIAS, int OUT_MODE>
__global__ void __launch_bounds__(256) gemm_bf16_wmma(const unsigned short* __restrict__ A,
                                                      const unsigned short* __restrict__ B,
                                                      const float* __restrict__ bias,
                                                      void* __restrict__ Cv,
                                                      int M, int Ncols, int K, int ldc) {
  (void)M; (void)Ncols;
  constexpr int LS  = 40;        // LDS row stride in halves (32 data + 8 pad, 80B rows)
  constexpr int BUF = 128 * LS;  // halves per buffer per operand
  __shared__ __align__(16) unsigned short lA[2 * BUF];
  __shared__ __align__(16) unsigned short lB[2 * BUF];

  const int t     = threadIdx.x;
  const int lane  = t & 31;
  const int wave  = t >> 5;            // 0..7
  const int wm0   = (wave & 3) * 32;   // 4 waves along M
  const int wn0   = (wave >> 2) * 64;  // 2 waves along N
  const int mBase = blockIdx.y * 128;
  const int nBase = blockIdx.x * 128;

  const int lrow = t >> 2;          // 0..63
  const int lcol = (t & 3) * 8;     // 0,8,16,24 (halves)
  const int st0  = lrow * LS + lcol;
  const int st1  = (64 + lrow) * LS + lcol;

  const v8f vz = {0.f, 0.f, 0.f, 0.f, 0.f, 0.f, 0.f, 0.f};
  v8f acc[2][4] = {{vz, vz, vz, vz}, {vz, vz, vz, vz}};

  const unsigned short* Ag0 = A + (size_t)(mBase + lrow) * K + lcol;
  const unsigned short* Ag1 = Ag0 + (size_t)64 * K;
  const unsigned short* Bg0 = B + (size_t)(nBase + lrow) * K + lcol;
  const unsigned short* Bg1 = Bg0 + (size_t)64 * K;

  // prologue: stage K-tile 0 into buffer 0
  uint4 ra0 = *(const uint4*)(Ag0);
  uint4 ra1 = *(const uint4*)(Ag1);
  uint4 rb0 = *(const uint4*)(Bg0);
  uint4 rb1 = *(const uint4*)(Bg1);
  *(uint4*)&lA[st0] = ra0;
  *(uint4*)&lA[st1] = ra1;
  *(uint4*)&lB[st0] = rb0;
  *(uint4*)&lB[st1] = rb1;
  __syncthreads();

  int bufOff = 0;
  for (int kb = 0; kb < K; kb += 32) {
    const bool hasNext = (kb + 32) < K;
    if (hasNext) {  // issue next K-tile global loads; they overlap the WMMAs below
      ra0 = *(const uint4*)(Ag0 + kb + 32);
      ra1 = *(const uint4*)(Ag1 + kb + 32);
      rb0 = *(const uint4*)(Bg0 + kb + 32);
      rb1 = *(const uint4*)(Bg1 + kb + 32);
      if (kb + 64 < K) {  // hint the tile after that -> global_prefetch_b8
        __builtin_prefetch(Ag0 + kb + 64, 0, 1);
        __builtin_prefetch(Ag1 + kb + 64, 0, 1);
        __builtin_prefetch(Bg0 + kb + 64, 0, 1);
        __builtin_prefetch(Bg1 + kb + 64, 0, 1);
      }
    }

    // fragments per CDNA5 ISA 16-bit layouts
    const int ar  = lane & 15;
    const int ak  = (lane >> 4) * 8;   // A: K runs {ak..ak+7, ak+16..ak+23}
    const int bk2 = (lane >> 4) * 8;   // B: K runs {bk2..bk2+7, bk2+16..bk2+23}
    FragBf fa[2], fb[4];
#pragma unroll
    for (int mt = 0; mt < 2; ++mt) {
      const unsigned short* p = &lA[bufOff + (wm0 + mt * 16 + ar) * LS];
      fa[mt].q[0] = *(const uint4*)(p + ak);
      fa[mt].q[1] = *(const uint4*)(p + ak + 16);
    }
#pragma unroll
    for (int nt = 0; nt < 4; ++nt) {
      const unsigned short* p = &lB[bufOff + (wn0 + nt * 16 + ar) * LS + bk2];
      fb[nt].q[0] = *(const uint4*)(p);
      fb[nt].q[1] = *(const uint4*)(p + 16);
    }
#pragma unroll
    for (int mt = 0; mt < 2; ++mt)
#pragma unroll
      for (int nt = 0; nt < 4; ++nt)
        acc[mt][nt] = __builtin_amdgcn_wmma_f32_16x16x32_bf16(
            false, fa[mt].v, false, fb[nt].v, (short)0, acc[mt][nt], false, false);

    if (hasNext) {  // stage next tile into the other buffer; single barrier per iter
      const int nb = bufOff ^ BUF;
      *(uint4*)&lA[nb + st0] = ra0;
      *(uint4*)&lA[nb + st1] = ra1;
      *(uint4*)&lB[nb + st0] = rb0;
      *(uint4*)&lB[nb + st1] = rb1;
      __syncthreads();
      bufOff = nb;
    }
  }

  // epilogue: C/D layout: M = r + (lane>=16)*8, N = lane&15
#pragma unroll
  for (int nt = 0; nt < 4; ++nt) {
    const int n = nBase + wn0 + nt * 16 + (lane & 15);
    float bb = 0.f;
    if constexpr (ADD_BIAS) bb = bias[n];
#pragma unroll
    for (int mt = 0; mt < 2; ++mt) {
      const int m0 = mBase + wm0 + mt * 16 + ((lane >> 4) << 3);
      if constexpr (OUT_MODE == 2) {
        float* C = (float*)Cv;
#pragma unroll
        for (int r = 0; r < 8; ++r)
          C[(size_t)(m0 + r) * ldc + n] = acc[mt][nt][r] + bb;
      } else if constexpr (OUT_MODE == 0) {
        unsigned short* C = (unsigned short*)Cv;
#pragma unroll
        for (int r = 0; r < 8; ++r)
          C[(size_t)(m0 + r) * ldc + n] = f32_to_bf16_rne(acc[mt][nt][r] + bb);
      } else {  // transposed bf16: 8 consecutive m per lane -> one 16B store
        unsigned short* C = (unsigned short*)Cv;
        unsigned short h[8];
#pragma unroll
        for (int r = 0; r < 8; ++r) h[r] = f32_to_bf16_rne(acc[mt][nt][r] + bb);
        uint4 pk;
        pk.x = (unsigned)h[0] | ((unsigned)h[1] << 16);
        pk.y = (unsigned)h[2] | ((unsigned)h[3] << 16);
        pk.z = (unsigned)h[4] | ((unsigned)h[5] << 16);
        pk.w = (unsigned)h[6] | ((unsigned)h[7] << 16);
        *(uint4*)&C[(size_t)n * ldc + m0] = pk;
      }
    }
  }
}

// ---------------------------------------------------------------------------
// Per-row: w = softmax(s / max(||s||_2, eps)), output bf16.
// One block (256 thr) per row of 8192 f32; row kept in registers (32 f32/thr).
// softmax(s/norm) computed as exp((s - smax)/norm) / sum.
// ---------------------------------------------------------------------------
__global__ void __launch_bounds__(256) row_l2norm_softmax_bf16(const float* __restrict__ s,
                                                               unsigned short* __restrict__ w) {
  const int row = blockIdx.x;
  const int t   = threadIdx.x;
  const float4* s4 = (const float4*)(s + (size_t)row * Nn);

  float4 r[8];
  float sumsq = 0.f;
  float mx = -3.402823466e38f;
#pragma unroll
  for (int i = 0; i < 8; ++i) {
    r[i] = s4[t + i * 256];
    sumsq += r[i].x * r[i].x + r[i].y * r[i].y + r[i].z * r[i].z + r[i].w * r[i].w;
    mx = fmaxf(mx, fmaxf(fmaxf(r[i].x, r[i].y), fmaxf(r[i].z, r[i].w)));
  }

  __shared__ float red[256];
  red[t] = mx; __syncthreads();
  for (int o = 128; o > 0; o >>= 1) { if (t < o) red[t] = fmaxf(red[t], red[t + o]); __syncthreads(); }
  mx = red[0]; __syncthreads();

  red[t] = sumsq; __syncthreads();
  for (int o = 128; o > 0; o >>= 1) { if (t < o) red[t] += red[t + o]; __syncthreads(); }
  const float inv = 1.0f / fmaxf(sqrtf(red[0]), 1e-12f);
  __syncthreads();

  float lsum = 0.f;
#pragma unroll
  for (int i = 0; i < 8; ++i) {
    r[i].x = __expf((r[i].x - mx) * inv);
    r[i].y = __expf((r[i].y - mx) * inv);
    r[i].z = __expf((r[i].z - mx) * inv);
    r[i].w = __expf((r[i].w - mx) * inv);
    lsum += r[i].x + r[i].y + r[i].z + r[i].w;
  }
  red[t] = lsum; __syncthreads();
  for (int o = 128; o > 0; o >>= 1) { if (t < o) red[t] += red[t + o]; __syncthreads(); }
  const float rs = 1.0f / red[0];

  unsigned long long* w8 = (unsigned long long*)(w + (size_t)row * Nn);
#pragma unroll
  for (int i = 0; i < 8; ++i) {
    unsigned long long p =
        (unsigned long long)f32_to_bf16_rne(r[i].x * rs) |
        ((unsigned long long)f32_to_bf16_rne(r[i].y * rs) << 16) |
        ((unsigned long long)f32_to_bf16_rne(r[i].z * rs) << 32) |
        ((unsigned long long)f32_to_bf16_rne(r[i].w * rs) << 48);
    w8[t + i * 256] = p;
  }
}

// ---------------------------------------------------------------------------
extern "C" void kernel_launch(void* const* d_in, const int* in_sizes, int n_in,
                              void* d_out, int out_size, void* d_ws, size_t ws_size,
                              hipStream_t stream) {
  (void)in_sizes; (void)n_in; (void)out_size; (void)ws_size;
  const float* x  = (const float*)d_in[0];
  const float* Wq = (const float*)d_in[1];
  const float* bq = (const float*)d_in[2];
  const float* Wk = (const float*)d_in[3];
  const float* bk = (const float*)d_in[4];
  const float* Wv = (const float*)d_in[5];
  const float* bv = (const float*)d_in[6];

  size_t off = 0;
  auto carve = [&](size_t bytes) -> void* {
    void* p = (char*)d_ws + off;
    off += (bytes + 255) & ~(size_t)255;
    return p;
  };
  unsigned short* x_bf  = (unsigned short*)carve((size_t)Nn * Ee * 2);
  unsigned short* wq_bf = (unsigned short*)carve((size_t)Ee * Ee * 2);
  unsigned short* wk_bf = (unsigned short*)carve((size_t)Ee * Ee * 2);
  unsigned short* wv_bf = (unsigned short*)carve((size_t)Ee * Ee * 2);
  unsigned short* q_bf  = (unsigned short*)carve((size_t)Nn * Ee * 2);
  unsigned short* k_bf  = (unsigned short*)carve((size_t)Nn * Ee * 2);
  unsigned short* vT_bf = (unsigned short*)carve((size_t)Ee * Nn * 2);  // v^T [E][N]
  float*          s_f   = (float*)carve((size_t)Nn * Nn * 4);
  unsigned short* w_bf  = (unsigned short*)carve((size_t)Nn * Nn * 2);

  // fp32 -> bf16 operand rounding (accumulation stays fp32 inside WMMA)
  cvt_f32_to_bf16<<<(Nn * Ee / 4 + 255) / 256, 256, 0, stream>>>(x, x_bf, Nn * Ee / 4);
  cvt_f32_to_bf16<<<(Ee * Ee / 4 + 255) / 256, 256, 0, stream>>>(Wq, wq_bf, Ee * Ee / 4);
  cvt_f32_to_bf16<<<(Ee * Ee / 4 + 255) / 256, 256, 0, stream>>>(Wk, wk_bf, Ee * Ee / 4);
  cvt_f32_to_bf16<<<(Ee * Ee / 4 + 255) / 256, 256, 0, stream>>>(Wv, wv_bf, Ee * Ee / 4);

  // q,k,v projections: y = x @ W^T + b  (A=x rows, B=W rows -> A*B^T)
  dim3 gQKV(Ee / 128, Nn / 128);
  gemm_bf16_wmma<true, 0><<<gQKV, 256, 0, stream>>>(x_bf, wq_bf, bq, q_bf, Nn, Ee, Ee, Ee);
  gemm_bf16_wmma<true, 0><<<gQKV, 256, 0, stream>>>(x_bf, wk_bf, bk, k_bf, Nn, Ee, Ee, Ee);
  gemm_bf16_wmma<true, 1><<<gQKV, 256, 0, stream>>>(x_bf, wv_bf, bv, vT_bf, Nn, Ee, Ee, Nn);

  // s = q @ k^T  (fp32 out, streams through L2 -> HBM)
  dim3 gS(Nn / 128, Nn / 128);
  gemm_bf16_wmma<false, 2><<<gS, 256, 0, stream>>>(q_bf, k_bf, nullptr, s_f, Nn, Nn, Ee, Nn);

  // w = softmax(s / max(||s||, eps)) -> bf16 (halves the dominant read stream)
  row_l2norm_softmax_bf16<<<Nn, 256, 0, stream>>>(s_f, w_bf);

  // out = w @ v = w @ (vT)^T  (fp32 out to d_out)
  dim3 gO(Ee / 128, Nn / 128);
  gemm_bf16_wmma<false, 2><<<gO, 256, 0, stream>>>(w_bf, vT_bf, nullptr, (float*)d_out,
                                                   Nn, Ee, Nn, Ee);
}
